// QuantumHybrid_41377714929711
// MI455X (gfx1250) — matrix operand order repaired and checked by reading, not compiled, not run on hardware
//
#include <hip/hip_runtime.h>
#include <math.h>

typedef __attribute__((ext_vector_type(16))) __bf16 v16bf;
typedef __attribute__((ext_vector_type(8)))  float  v8f;

#define D_IN   1280
#define H_DIM  64
#define BATCH  32768
#define N_CLS  10
#define BN_EPS 1e-5f

#define WMMA_BF16(Acc, A, B) \
  __builtin_amdgcn_wmma_f32_16x16x32_bf16(false, (A), false, (B), (short)0, (Acc), false, false)

// ---------------------------------------------------------------------------
// Kernel 0: split W1 (fp32, [64,1280]) into bf16 hi + bf16 lo (hi+lo ~ fp32)
// ---------------------------------------------------------------------------
__global__ __launch_bounds__(256) void k_split_w1(const float* __restrict__ W1,
                                                  __bf16* __restrict__ w1h,
                                                  __bf16* __restrict__ w1l) {
  int i = blockIdx.x * blockDim.x + threadIdx.x;
  if (i < H_DIM * D_IN) {
    float f  = W1[i];
    __bf16 h = (__bf16)f;
    w1h[i] = h;
    w1l[i] = (__bf16)(f - (float)h);
  }
}

// ---------------------------------------------------------------------------
// Build one 16-bit A fragment pair (hi/lo) for WMMA A in this code's contiguous k order (B uses the same):
// lane<16: M=lane, K in {0..15}; lane>=16: M=lane-16, K+16.
// ---------------------------------------------------------------------------
__device__ __forceinline__ void load_a_frag(const float* __restrict__ p,
                                            v16bf& ah, v16bf& al) {
  const float4* p0 = reinterpret_cast<const float4*>(p);
  const float4* p1 = reinterpret_cast<const float4*>(p + 8);
  float4 a0 = p0[0], a1 = p0[1], a2 = p1[0], a3 = p1[1];
  float af[16] = {a0.x, a0.y, a0.z, a0.w, a1.x, a1.y, a1.z, a1.w,
                  a2.x, a2.y, a2.z, a2.w, a3.x, a3.y, a3.z, a3.w};
#pragma unroll
  for (int j = 0; j < 16; ++j) {
    __bf16 hb = (__bf16)af[j];
    ah[j] = hb;
    al[j] = (__bf16)(af[j] - (float)hb);
  }
}

// ---------------------------------------------------------------------------
// Kernel 1: fused  h=relu(x@W1^T+b1) -> BN -> xq=tanh(h@W2^T+b2)
// One wave computes a 32x64 output tile (2 M-subtiles x 4 N-tiles),
// 24 v_wmma_f32_16x16x32_bf16 per K-step of 32.  4 waves/block, 256 blocks.
// ---------------------------------------------------------------------------
__global__ __launch_bounds__(128) void k_front(
    const float* __restrict__ x,
    const __bf16* __restrict__ w1h, const __bf16* __restrict__ w1l,
    const float* __restrict__ b1,
    const float* __restrict__ bn_g, const float* __restrict__ bn_b,
    const float* __restrict__ bn_m, const float* __restrict__ bn_v,
    const float* __restrict__ W2,  const float* __restrict__ b2,
    float* __restrict__ xq) {
  __shared__ float hs[4][32][H_DIM + 1];

  const int wave  = threadIdx.x >> 5;
  const int lane  = threadIdx.x & 31;
  const int half  = lane >> 4;          // 0: lanes 0-15, 1: lanes 16-31
  const int lan16 = lane & 15;
  const int m0    = (blockIdx.x * 4 + wave) * 32;

  // Per-lane loop-invariant base pointers; tiles become immediate offsets.
  const int g0 = half * 16;
  const float* xr0 = x + (size_t)(m0 + lan16) * D_IN + g0;
  const float* xr1 = x + (size_t)(m0 + 16 + lan16) * D_IN + g0;
  const char*  pbh = (const char*)(w1h + (size_t)lan16 * D_IN + half * 16);
  const char*  pbl = (const char*)(w1l + (size_t)lan16 * D_IN + half * 16);
  constexpr int TS = 16 * D_IN * 2;     // 40960 B: N-tile stride (fits IOFFSET)

  v8f acc0[4] = {};
  v8f acc1[4] = {};

  for (int kb = 0; kb < D_IN; kb += 32) {
    if (kb + 32 < D_IN) {
      __builtin_prefetch(xr0 + kb + 32, 0, 1);     // global_prefetch_b8
      __builtin_prefetch(xr1 + kb + 32, 0, 1);
    }

    // ---- A fragments (fp32 -> bf16 hi/lo), 2 M-subtiles ----
    v16bf ah0, al0, ah1, al1;
    load_a_frag(xr0 + kb, ah0, al0);
    load_a_frag(xr1 + kb, ah1, al1);

    // ---- all 8 B fragments up front (distinct regs -> no WAR stalls) ----
    const int ko = kb * 2;
    v16bf bh0 = *(const v16bf*)(pbh + ko);
    v16bf bh1 = *(const v16bf*)(pbh + ko + TS);
    v16bf bh2 = *(const v16bf*)(pbh + ko + 2 * TS);
    v16bf bh3 = *(const v16bf*)(pbh + ko + 3 * TS);
    v16bf bl0 = *(const v16bf*)(pbl + ko);
    v16bf bl1 = *(const v16bf*)(pbl + ko + TS);
    v16bf bl2 = *(const v16bf*)(pbl + ko + 2 * TS);
    v16bf bl3 = *(const v16bf*)(pbl + ko + 3 * TS);

    // ---- 24 WMMAs, round-robin over 8 independent accumulators ----
    acc0[0] = WMMA_BF16(acc0[0], ah0, bh0);
    acc1[0] = WMMA_BF16(acc1[0], ah1, bh0);
    acc0[1] = WMMA_BF16(acc0[1], ah0, bh1);
    acc1[1] = WMMA_BF16(acc1[1], ah1, bh1);
    acc0[2] = WMMA_BF16(acc0[2], ah0, bh2);
    acc1[2] = WMMA_BF16(acc1[2], ah1, bh2);
    acc0[3] = WMMA_BF16(acc0[3], ah0, bh3);
    acc1[3] = WMMA_BF16(acc1[3], ah1, bh3);

    acc0[0] = WMMA_BF16(acc0[0], ah0, bl0);
    acc1[0] = WMMA_BF16(acc1[0], ah1, bl0);
    acc0[1] = WMMA_BF16(acc0[1], ah0, bl1);
    acc1[1] = WMMA_BF16(acc1[1], ah1, bl1);
    acc0[2] = WMMA_BF16(acc0[2], ah0, bl2);
    acc1[2] = WMMA_BF16(acc1[2], ah1, bl2);
    acc0[3] = WMMA_BF16(acc0[3], ah0, bl3);
    acc1[3] = WMMA_BF16(acc1[3], ah1, bl3);

    acc0[0] = WMMA_BF16(acc0[0], al0, bh0);
    acc1[0] = WMMA_BF16(acc1[0], al1, bh0);
    acc0[1] = WMMA_BF16(acc0[1], al0, bh1);
    acc1[1] = WMMA_BF16(acc1[1], al1, bh1);
    acc0[2] = WMMA_BF16(acc0[2], al0, bh2);
    acc1[2] = WMMA_BF16(acc1[2], al1, bh2);
    acc0[3] = WMMA_BF16(acc0[3], al0, bh3);
    acc1[3] = WMMA_BF16(acc1[3], al1, bh3);
  }

  // ---- epilogue: bias + relu + batchnorm, spill to LDS ----
  // C/D layout: VGPR r holds (M = r + 8*half, N = lane%16) per tile.
#pragma unroll
  for (int t = 0; t < 4; ++t) {
    const int n = t * 16 + lan16;
    const float bias = b1[n];
    const float gg = bn_g[n], bb = bn_b[n], mu = bn_m[n];
    const float inv = rsqrtf(bn_v[n] + BN_EPS);
#pragma unroll
    for (int r = 0; r < 8; ++r) {
      float h0 = fmaxf(acc0[t][r] + bias, 0.0f);
      hs[wave][r + half * 8][n] = (h0 - mu) * inv * gg + bb;
      float h1 = fmaxf(acc1[t][r] + bias, 0.0f);
      hs[wave][16 + r + half * 8][n] = (h1 - mu) * inv * gg + bb;
    }
  }
  __syncthreads();

  // ---- tiny second GEMM: xq[32 rows][4] = tanh(h @ W2^T + b2) ----
#pragma unroll
  for (int o = lane; o < 128; o += 32) {
    const int row = o >> 2, q = o & 3;
    float s = b2[q];
    const float* w2r = W2 + q * H_DIM;
#pragma unroll 8
    for (int n = 0; n < H_DIM; ++n) s += hs[wave][row][n] * w2r[n];
    xq[(size_t)(m0 + row) * 4 + q] = tanhf(s);
  }
}

// ---------------------------------------------------------------------------
// Kernel 2: 4-qubit statevector sim per row + final linear (4 -> 10)
// Wire i maps to bit (3-i) of the flattened index.
// ---------------------------------------------------------------------------
__device__ __forceinline__ void apply_ry(float* sr, float* si, float half, int mask) {
  float c = cosf(half), s = sinf(half);
#pragma unroll
  for (int idx = 0; idx < 16; ++idx) {
    if (!(idx & mask)) {
      int j = idx | mask;
      float r0 = sr[idx], i0 = si[idx], r1 = sr[j], i1 = si[j];
      sr[idx] = c * r0 - s * r1;  si[idx] = c * i0 - s * i1;
      sr[j]   = s * r0 + c * r1;  si[j]   = s * i0 + c * i1;
    }
  }
}

__device__ __forceinline__ void apply_rz(float* sr, float* si, float half, int mask) {
  float c = cosf(half), s = sinf(half);
#pragma unroll
  for (int idx = 0; idx < 16; ++idx) {
    float r = sr[idx], i = si[idx];
    if (idx & mask) { sr[idx] = c * r - s * i; si[idx] = c * i + s * r; }  // *conj(p)
    else           { sr[idx] = c * r + s * i; si[idx] = c * i - s * r; }  // *p
  }
}

__device__ __forceinline__ void apply_cnot(float* sr, float* si, int mc, int mt) {
#pragma unroll
  for (int idx = 0; idx < 16; ++idx) {
    if ((idx & mc) && !(idx & mt)) {
      int j = idx | mt;
      float tr = sr[idx], ti = si[idx];
      sr[idx] = sr[j]; si[idx] = si[j];
      sr[j] = tr;      si[j] = ti;
    }
  }
}

__global__ __launch_bounds__(256) void k_quantum(
    const float* __restrict__ xq, const float* __restrict__ qw,
    const float* __restrict__ W3, const float* __restrict__ b3,
    float* __restrict__ out) {
  int m = blockIdx.x * blockDim.x + threadIdx.x;
  if (m >= BATCH) return;

  const float PI = 3.14159265358979323846f;
  float sr[16], si[16];
#pragma unroll
  for (int i = 0; i < 16; ++i) { sr[i] = 0.f; si[i] = 0.f; }
  sr[0] = 1.f;

#pragma unroll
  for (int w = 0; w < 4; ++w) {
    float ang = tanhf(xq[(size_t)m * 4 + w]) * PI;   // reference re-tanh's input
    apply_ry(sr, si, ang * 0.5f, 1 << (3 - w));
  }

#pragma unroll
  for (int l = 0; l < 2; ++l) {
#pragma unroll
    for (int i = 0; i < 4; ++i) {
      apply_ry(sr, si, qw[l * 8 + i * 2 + 0] * 0.5f, 1 << (3 - i));
      apply_rz(sr, si, qw[l * 8 + i * 2 + 1] * 0.5f, 1 << (3 - i));
    }
#pragma unroll
    for (int i = 0; i < 4; ++i)
      apply_cnot(sr, si, 1 << (3 - i), 1 << (3 - ((i + 1) & 3)));
  }

  float z[4];
#pragma unroll
  for (int i = 0; i < 4; ++i) {
    float acc = 0.f;
#pragma unroll
    for (int idx = 0; idx < 16; ++idx) {
      float p = sr[idx] * sr[idx] + si[idx] * si[idx];
      acc += ((idx >> (3 - i)) & 1) ? -p : p;
    }
    z[i] = acc;
  }

#pragma unroll
  for (int c = 0; c < N_CLS; ++c) {
    float s = b3[c];
#pragma unroll
    for (int q = 0; q < 4; ++q) s += z[q] * W3[c * 4 + q];
    out[(size_t)m * N_CLS + c] = s;
  }
}

// ---------------------------------------------------------------------------
extern "C" void kernel_launch(void* const* d_in, const int* in_sizes, int n_in,
                              void* d_out, int out_size, void* d_ws, size_t ws_size,
                              hipStream_t stream) {
  const float* x    = (const float*)d_in[0];
  const float* W1   = (const float*)d_in[1];
  const float* b1   = (const float*)d_in[2];
  const float* bn_g = (const float*)d_in[3];
  const float* bn_b = (const float*)d_in[4];
  const float* bn_m = (const float*)d_in[5];
  const float* bn_v = (const float*)d_in[6];
  const float* W2   = (const float*)d_in[7];
  const float* b2   = (const float*)d_in[8];
  const float* qw   = (const float*)d_in[9];
  const float* W3   = (const float*)d_in[10];
  const float* b3   = (const float*)d_in[11];
  float* out = (float*)d_out;

  char* ws = (char*)d_ws;
  __bf16* w1h = (__bf16*)ws;                                       // 160 KB
  __bf16* w1l = (__bf16*)(ws + (size_t)H_DIM * D_IN * 2);          // 160 KB
  float*  xq  = (float*)(ws + (size_t)H_DIM * D_IN * 4);           // 512 KB

  k_split_w1<<<(H_DIM * D_IN + 255) / 256, 256, 0, stream>>>(W1, w1h, w1l);
  k_front<<<BATCH / 128, 128, 0, stream>>>(x, w1h, w1l, b1, bn_g, bn_b,
                                           bn_m, bn_v, W2, b2, xq);
  k_quantum<<<BATCH / 256, 256, 0, stream>>>(xq, qw, W3, b3, out);
}
